// MultiModalSpikeTransformer_82463372083297
// MI455X (gfx1250) — compile-verified, matrix operand order checked by reading, not run on hardware
//
#include <hip/hip_runtime.h>

// ============================================================================
// MultiModalSpikeTransformer for MI455X (gfx1250), wave32 + WMMA.
//
// Strategy:
//  * All GEMMs run on v_wmma_f32_16x16x32_bf16 (fp32 accum). Weights are
//    converted fp32 -> bf16 *transposed* ([N,K]) once per call so A and B
//    fragments load identically from LDS as two contiguous 16B runs per lane
//    (matches the ISA 16-bit A-matrix 16x32 VGPR layout).
//  * LIF (32 steps, threshold 1, soft reset, rate code) is exact and fused
//    into the GEMM epilogue where the reference applies it post-linear.
//  * Transformer-layer attention has seq-len 1: softmax over one logit == 1,
//    so attention output == o_proj(v_proj(x)) exactly; q/k projections skipped.
//  * Cross-attention core (64 queries x 64 keys x dh=64, 1024 blocks) is a
//    small LDS kernel (fp32), ~1 GFLOP total — negligible next to the
//    [8192,512]x[512,512] WMMA projections.
// ============================================================================

typedef __bf16 bf16;
typedef __attribute__((ext_vector_type(16))) __bf16 bf16x16;
typedef __attribute__((ext_vector_type(8)))  __bf16 bf16x8;
typedef __attribute__((ext_vector_type(8)))  float  f32x8;

#define TSTEPS 32

// ---------- device helpers ----------
__device__ __forceinline__ float lif_rate(float x) {
  // LIF with constant drive x: mem += x; s = (mem-1 > 0); mem -= s; rate = mean(s)
  float mem = 0.0f, cnt = 0.0f;
#pragma unroll
  for (int t = 0; t < TSTEPS; ++t) {
    mem += x;
    float s = (mem - 1.0f > 0.0f) ? 1.0f : 0.0f;
    mem -= s;
    cnt += s;
  }
  return cnt * (1.0f / (float)TSTEPS);
}

__device__ __forceinline__ bf16 f2bf(float f) {
  unsigned u = __builtin_bit_cast(unsigned, f);
  unsigned r = u + 0x7FFFu + ((u >> 16) & 1u);   // round to nearest even
  unsigned short h = (unsigned short)(r >> 16);
  return __builtin_bit_cast(bf16, h);
}

// ---------- conversion kernels ----------
__global__ void k_cvt(const float* __restrict__ src, bf16* __restrict__ dst, int n) {
  int i = blockIdx.x * 256 + threadIdx.x;
  if (i < n) dst[i] = f2bf(src[i]);
}

// src [R,C] fp32 row-major  ->  dst [C,R] bf16 row-major (i.e. W^T)
__global__ void k_cvt_t(const float* __restrict__ src, bf16* __restrict__ dst,
                        int R, int C) {
  int i = blockIdx.x * 256 + threadIdx.x;
  if (i >= R * C) return;
  int r = i / C, c = i - r * C;
  dst[(size_t)c * R + r] = f2bf(src[i]);
}

// ---------- WMMA GEMM: C[M,N] = A[M,K](bf16) @ Wt[N,K](bf16)^T + bias, epi ----
#define BM 128
#define BN 64
#define BK 32
#define LDSA 40   // padded row stride (bf16 elems) -> 80B rows, 16B aligned runs
#define LDSB 40

__global__ __launch_bounds__(256)
void k_gemm(const bf16* __restrict__ A, const bf16* __restrict__ Bt,
            const float* __restrict__ bias, float* __restrict__ C,
            int M, int N, int K, int epi) {
  __shared__ __attribute__((aligned(16))) bf16 sA[BM * LDSA];
  __shared__ __attribute__((aligned(16))) bf16 sB[BN * LDSB];

  const int tid  = threadIdx.x;
  const int lane = tid & 31;
  const int wave = tid >> 5;        // 0..7
  const int wr   = wave >> 1;       // 0..3  (M direction, 32 rows each)
  const int wc   = wave & 1;        // 0..1  (N direction, 32 cols each)
  const int nIn  = lane & 15;       // row (A) / col (B) within 16-tile
  const int half = lane >> 4;       // K-half selector per ISA layout
  const int m0   = blockIdx.y * BM;
  const int n0   = blockIdx.x * BN;

  f32x8 acc[2][2] = {};

  for (int k0 = 0; k0 < K; k0 += BK) {
    // ---- cooperative A tile load: 128x32 bf16, 16 elems (32B) per thread ----
    {
      int r = tid >> 1;
      int c = (tid & 1) << 4;
      int gm = m0 + r;
      uint4 v0 = make_uint4(0, 0, 0, 0), v1 = v0;
      if (gm < M) {
        const uint4* gp = (const uint4*)(A + (size_t)gm * K + k0 + c);
        v0 = gp[0];
        v1 = gp[1];
        if (k0 + BK < K)  // speculative prefetch of next K tile
          __builtin_prefetch(A + (size_t)gm * K + k0 + BK + c, 0, 1);
      }
      *(uint4*)(&sA[r * LDSA + c])     = v0;
      *(uint4*)(&sA[r * LDSA + c + 8]) = v1;
    }
    // ---- cooperative B tile load: 64x32 bf16 (rows of W^T), 8 elems each ----
    {
      int r = tid >> 2;
      int c = (tid & 3) << 3;
      int gn = n0 + r;
      uint4 v = make_uint4(0, 0, 0, 0);
      if (gn < N) {
        v = *(const uint4*)(Bt + (size_t)gn * K + k0 + c);
        if (k0 + BK < K)
          __builtin_prefetch(Bt + (size_t)gn * K + k0 + BK + c, 0, 1);
      }
      *(uint4*)(&sB[r * LDSB + c]) = v;
    }
    __syncthreads();

    // ---- build fragments (two contiguous 16B runs per lane) and WMMA ----
    bf16x16 af[2], bfr[2];
#pragma unroll
    for (int i = 0; i < 2; ++i) {
      {
        const bf16* p = &sA[(wr * 32 + i * 16 + nIn) * LDSA + half * 8];
        bf16x8 lo = *(const bf16x8*)p;           // k = half*8 .. +7
        bf16x8 hi = *(const bf16x8*)(p + 16);    // k = 16+half*8 .. +7
#pragma unroll
        for (int e = 0; e < 8; ++e) { af[i][e] = lo[e]; af[i][e + 8] = hi[e]; }
      }
      {
        const bf16* p = &sB[(wc * 32 + i * 16 + nIn) * LDSB + half * 8];
        bf16x8 lo = *(const bf16x8*)p;
        bf16x8 hi = *(const bf16x8*)(p + 16);
#pragma unroll
        for (int e = 0; e < 8; ++e) { bfr[i][e] = lo[e]; bfr[i][e + 8] = hi[e]; }
      }
    }
#pragma unroll
    for (int mi = 0; mi < 2; ++mi)
#pragma unroll
      for (int ni = 0; ni < 2; ++ni)
        acc[mi][ni] = __builtin_amdgcn_wmma_f32_16x16x32_bf16(
            false, af[mi], false, bfr[ni], (short)0, acc[mi][ni], false, false);
    __syncthreads();
  }

  // ---- epilogue: bias (+ optional LIF), guarded store ----
#pragma unroll
  for (int mi = 0; mi < 2; ++mi)
#pragma unroll
    for (int ni = 0; ni < 2; ++ni) {
      int nG = n0 + wc * 32 + ni * 16 + nIn;
#pragma unroll
      for (int e = 0; e < 8; ++e) {
        int mG = m0 + wr * 32 + mi * 16 + half * 8 + e;
        if (mG < M && nG < N) {
          float v = acc[mi][ni][e] + (bias ? bias[nG] : 0.0f);
          if (epi == 1) v = lif_rate(v);
          C[(size_t)mG * N + nG] = v;
        }
      }
    }
}

// ---------- cross-attention core: per (bt, head), S=64, dh=64, D=512 --------
__global__ __launch_bounds__(64)
void k_attn(const float* __restrict__ Q, const float* __restrict__ K,
            const float* __restrict__ V, float* __restrict__ O) {
  __shared__ float sQ[64 * 65], sK[64 * 65], sV[64 * 65], sP[64 * 65];
  const int bt = blockIdx.x, h = blockIdx.y, t = threadIdx.x;
  const size_t base = ((size_t)bt * 64 + t) * 512 + h * 64;
#pragma unroll 8
  for (int c = 0; c < 64; ++c) {
    sQ[t * 65 + c] = Q[base + c];
    sK[t * 65 + c] = K[base + c];
    sV[t * 65 + c] = V[base + c];
  }
  __syncthreads();
  float mx = -3.0e38f;
  for (int k = 0; k < 64; ++k) {
    float d = 0.0f;
#pragma unroll 16
    for (int e = 0; e < 64; ++e) d += sQ[t * 65 + e] * sK[k * 65 + e];
    d *= 0.125f;                           // dh^-0.5
    sP[t * 65 + k] = d;
    mx = fmaxf(mx, d);
  }
  float sum = 0.0f;
  for (int k = 0; k < 64; ++k) {
    float e = __expf(sP[t * 65 + k] - mx);
    sP[t * 65 + k] = e;
    sum += e;
  }
  const float inv = 1.0f / sum;
  for (int d0 = 0; d0 < 64; ++d0) {
    float o = 0.0f;
#pragma unroll 16
    for (int k = 0; k < 64; ++k) o += sP[t * 65 + k] * sV[k * 65 + d0];
    O[base + d0] = o * inv;
  }
}

// ---------- mean over contiguous row groups -------------------------------
// in: [groups*S, C] -> out[g, outOff + c] = mean_s in[(g*S+s)*C + c]
__global__ void k_mean_rows(const float* __restrict__ in, float* __restrict__ out,
                            int groups, int S, int C, int outLD, int outOff) {
  int idx = blockIdx.x * 256 + threadIdx.x;
  if (idx >= groups * C) return;
  int g = idx / C, c = idx - g * C;
  const float* p = in + (size_t)g * S * C + c;
  float s = 0.0f;
  for (int k = 0; k < S; ++k) s += p[(size_t)k * C];
  out[(size_t)g * outLD + outOff + c] = s / (float)S;
}

// ---------- layernorm (optional residual, optional fused LIF) --------------
__global__ __launch_bounds__(256)
void k_layernorm(const float* __restrict__ x, const float* __restrict__ res,
                 const float* __restrict__ g, const float* __restrict__ b,
                 float* __restrict__ out, int C, int doLif) {
  __shared__ float red[256];
  const int row = blockIdx.x, tid = threadIdx.x;
  const float* xr = x + (size_t)row * C;
  const float* rr = res ? res + (size_t)row * C : nullptr;

  float s = 0.0f;
  for (int c = tid; c < C; c += 256) s += xr[c] + (rr ? rr[c] : 0.0f);
  red[tid] = s; __syncthreads();
  for (int o = 128; o > 0; o >>= 1) { if (tid < o) red[tid] += red[tid + o]; __syncthreads(); }
  const float mu = red[0] / (float)C;
  __syncthreads();

  float s2 = 0.0f;
  for (int c = tid; c < C; c += 256) {
    float v = xr[c] + (rr ? rr[c] : 0.0f) - mu;
    s2 += v * v;
  }
  red[tid] = s2; __syncthreads();
  for (int o = 128; o > 0; o >>= 1) { if (tid < o) red[tid] += red[tid + o]; __syncthreads(); }
  const float inv = 1.0f / sqrtf(red[0] / (float)C + 1e-5f);

  for (int c = tid; c < C; c += 256) {
    float v = (xr[c] + (rr ? rr[c] : 0.0f) - mu) * inv * g[c] + b[c];
    if (doLif) v = lif_rate(v);
    out[(size_t)row * C + c] = v;
  }
}

// ===========================================================================
extern "C" void kernel_launch(void* const* d_in, const int* in_sizes, int n_in,
                              void* d_out, int out_size, void* d_ws, size_t ws_size,
                              hipStream_t stream) {
  (void)in_sizes; (void)out_size; (void)ws_size;
  if (n_in < 118) return;  // expected flattening: vision, audio, params (DFS dict order)

  const float* vision = (const float*)d_in[0];
  const float* audio  = (const float*)d_in[1];
  auto P = [&](int i) -> const float* { return (const float*)d_in[i]; };

  const int Bn = 8, Tt = 16, Ss = 64, Dv = 768, Da = 512, D = 512;
  const int Rr = Bn * Tt * Ss;  // 8192 token rows for fusion stage
  const int BT = Bn * Tt;       // 128
  const int NC = 1000;

  // ---- workspace bump allocator (needs ~147 MB) ----
  char* cur = (char*)d_ws;
  auto alloc = [&](size_t bytes) -> void* {
    void* r = (void*)cur;
    cur += (bytes + 255) & ~(size_t)255;
    return r;
  };

  // bf16 transposed weights
  bf16* w_pv  = (bf16*)alloc((size_t)Dv * D * 2);
  bf16* w_pa  = (bf16*)alloc((size_t)D * D * 2);
  bf16* w_q   = (bf16*)alloc((size_t)D * D * 2);
  bf16* w_k   = (bf16*)alloc((size_t)D * D * 2);
  bf16* w_v   = (bf16*)alloc((size_t)D * D * 2);
  bf16* w_o   = (bf16*)alloc((size_t)D * D * 2);
  bf16* w_f1  = (bf16*)alloc((size_t)1024 * 1024 * 2);
  bf16* w_f2  = (bf16*)alloc((size_t)1024 * 512 * 2);
  bf16* w_cls = (bf16*)alloc((size_t)512 * 1000 * 2);
  bf16 *lw_v[6], *lw_o[6], *lw_f1[6], *lw_f2[6];
  for (int l = 0; l < 6; ++l) {
    lw_v[l]  = (bf16*)alloc((size_t)512 * 512 * 2);
    lw_o[l]  = (bf16*)alloc((size_t)512 * 512 * 2);
    lw_f1[l] = (bf16*)alloc((size_t)512 * 2048 * 2);
    lw_f2[l] = (bf16*)alloc((size_t)2048 * 512 * 2);
  }
  // activations
  bf16*  visb = (bf16*)alloc((size_t)Rr * Dv * 2);
  bf16*  audb = (bf16*)alloc((size_t)Rr * Da * 2);
  bf16*  pvb  = (bf16*)alloc((size_t)Rr * D * 2);
  bf16*  pab  = (bf16*)alloc((size_t)Rr * D * 2);
  float* F0   = (float*)alloc((size_t)Rr * D * 4);
  float* F1   = (float*)alloc((size_t)Rr * D * 4);
  float* F2   = (float*)alloc((size_t)Rr * D * 4);
  float* F3   = (float*)alloc((size_t)Rr * D * 4);
  bf16*  Hb   = audb;  // audio bf16 is dead after its projection; reuse
  float* concatB = (float*)alloc((size_t)BT * 1024 * 4);
  bf16*  smallb  = (bf16*)alloc((size_t)BT * 2048 * 2);
  float* S0 = (float*)alloc((size_t)BT * 2048 * 4);
  float* S1 = (float*)alloc((size_t)BT * 2048 * 4);
  float* X  = (float*)alloc((size_t)BT * D * 4);
  float* Xm = (float*)alloc((size_t)Bn * D * 4);

  auto gemm = [&](const bf16* Aa, const bf16* Bt, const float* bias, float* Cc,
                  int M, int N, int K, int epi) {
    dim3 g((unsigned)((N + BN - 1) / BN), (unsigned)((M + BM - 1) / BM));
    k_gemm<<<g, 256, 0, stream>>>(Aa, Bt, bias, Cc, M, N, K, epi);
  };
  auto cvt = [&](const float* s, bf16* d, size_t n) {
    k_cvt<<<(unsigned)((n + 255) / 256), 256, 0, stream>>>(s, d, (int)n);
  };
  auto cvtT = [&](const float* s, bf16* d, int R_, int C_) {
    size_t n = (size_t)R_ * C_;
    k_cvt_t<<<(unsigned)((n + 255) / 256), 256, 0, stream>>>(s, d, R_, C_);
  };
  auto lnorm = [&](const float* x, const float* res, const float* g,
                   const float* b, float* out, int doLif) {
    k_layernorm<<<BT, 256, 0, stream>>>(x, res, g, b, out, D, doLif);
  };
  auto attn = [&](const float* Q, const float* K, const float* V, float* O) {
    k_attn<<<dim3(BT, 8), 64, 0, stream>>>(Q, K, V, O);
  };

  // ---- weight conversion (fp32 -> bf16 transposed) ----
  cvtT(P(2),  w_pv, Dv, D);
  cvtT(P(4),  w_pa, D, D);
  cvtT(P(6),  w_q,  D, D);
  cvtT(P(8),  w_k,  D, D);
  cvtT(P(10), w_v,  D, D);
  cvtT(P(12), w_o,  D, D);
  cvtT(P(14), w_f1, 1024, 1024);
  cvtT(P(16), w_f2, 1024, 512);
  for (int l = 0; l < 6; ++l) {
    int b0 = 20 + 16 * l;
    cvtT(P(b0 + 4),  lw_v[l],  512, 512);
    cvtT(P(b0 + 6),  lw_o[l],  512, 512);
    cvtT(P(b0 + 10), lw_f1[l], 512, 2048);
    cvtT(P(b0 + 12), lw_f2[l], 2048, 512);
  }
  cvtT(P(116), w_cls, 512, 1000);

  // ---- fusion: modality projections ----
  cvt(vision, visb, (size_t)Rr * Dv);
  gemm(visb, w_pv, P(3), F0, Rr, D, Dv, 0);
  cvt(F0, pvb, (size_t)Rr * D);
  cvt(audio, audb, (size_t)Rr * Da);
  gemm(audb, w_pa, P(5), F0, Rr, D, Da, 0);
  cvt(F0, pab, (size_t)Rr * D);

  // ---- direction va: q=pv, k/v=pa (shared cross_va weights) ----
  gemm(pvb, w_q, P(7),  F0, Rr, D, D, 0);
  gemm(pab, w_k, P(9),  F1, Rr, D, D, 0);
  gemm(pab, w_v, P(11), F2, Rr, D, D, 0);
  attn(F0, F1, F2, F3);
  cvt(F3, Hb, (size_t)Rr * D);
  gemm(Hb, w_o, P(13), F0, Rr, D, D, 1);                 // o-proj + LIF
  k_mean_rows<<<(BT * D + 255) / 256, 256, 0, stream>>>(F0, concatB, BT, Ss, D, 1024, 0);

  // ---- direction av: q=pa, k/v=pv ----
  gemm(pab, w_q, P(7),  F0, Rr, D, D, 0);
  gemm(pvb, w_k, P(9),  F1, Rr, D, D, 0);
  gemm(pvb, w_v, P(11), F2, Rr, D, D, 0);
  attn(F0, F1, F2, F3);
  cvt(F3, Hb, (size_t)Rr * D);
  gemm(Hb, w_o, P(13), F0, Rr, D, D, 1);
  k_mean_rows<<<(BT * D + 255) / 256, 256, 0, stream>>>(F0, concatB, BT, Ss, D, 1024, D);

  // ---- fusion MLP + out_norm(+LIF) ----
  cvt(concatB, smallb, (size_t)BT * 1024);
  gemm(smallb, w_f1, P(15), S0, BT, 1024, 1024, 1);      // fc1 + LIF
  cvt(S0, smallb, (size_t)BT * 1024);
  gemm(smallb, w_f2, P(17), S1, BT, 512, 1024, 0);       // fc2
  lnorm(S1, nullptr, P(18), P(19), X, 1);                // lif(layernorm(fused))

  // ---- transformer layers (seq-len-1 attention == o(v(x)), exact) ----
  for (int l = 0; l < 6; ++l) {
    int b0 = 20 + 16 * l;
    cvt(X, smallb, (size_t)BT * D);
    gemm(smallb, lw_v[l], P(b0 + 5), S0, BT, D, D, 0);   // v-proj
    cvt(S0, smallb, (size_t)BT * D);
    gemm(smallb, lw_o[l], P(b0 + 7), S1, BT, D, D, 1);   // o-proj + LIF
    lnorm(S1, X, P(b0 + 8), P(b0 + 9), X, 0);            // x = LN(x + attn)
    cvt(X, smallb, (size_t)BT * D);
    gemm(smallb, lw_f1[l], P(b0 + 11), S0, BT, 2048, D, 1);  // fc1 + LIF
    cvt(S0, smallb, (size_t)BT * 2048);
    gemm(smallb, lw_f2[l], P(b0 + 13), S1, BT, D, 2048, 0);  // fc2
    lnorm(S1, X, P(b0 + 14), P(b0 + 15), X, 0);          // x = LN(x + ff)
  }

  // ---- classifier: mean over T then [8,512]x[512,1000] ----
  k_mean_rows<<<(Bn * D + 255) / 256, 256, 0, stream>>>(X, Xm, Bn, Tt, D, D, 0);
  cvt(Xm, smallb, (size_t)Bn * D);
  gemm(smallb, w_cls, P(117), (float*)d_out, Bn, NC, D, 0);
}
